// VGAEModel_17806934409354
// MI455X (gfx1250) — matrix operand order repaired and checked by reading, not compiled, not run on hardware
//
#include <hip/hip_runtime.h>

// ---------------------------------------------------------------------------
// VGAE forward: emb -> GraphConv(relu) -> {mean, log_std} GraphConv heads ->
// z = mean + noise*exp(log_std) -> sigmoid(z @ z^T)   [8192 x 8192 f32 out]
//
// - All dense GEMMs: v_wmma_f32_16x16x32_f16, register-blocked (4x4 for z@zT).
// - Edge aggregation: dst-sorted counting sort (int atomics + LDS scan) then
//   atomic-free gather-sum per destination (all reads L2-resident).
// - 268MB adjacency output written with nontemporal stores.
// ---------------------------------------------------------------------------

typedef __attribute__((ext_vector_type(16))) _Float16 v16h;
typedef __attribute__((ext_vector_type(8)))  _Float16 v8h;
typedef __attribute__((ext_vector_type(8)))  float    v8f;

#define NN      8192
#define EDGES   524288
#define INDIM   512
#define HID1    512
#define HID2    256

// ---------------------------------------------------------------------------
// Register-blocked f16 WMMA GEMM: C[M x Nout] = A[M x K] * Bt[Nout x K]^T.
// 4 waves per block in a 2x2 arrangement; each wave owns TM x TN 16x16 tiles.
// ACT: 0 = plain store, 1 = sigmoid + nontemporal store.
// Fragment layouts per CDNA5 ISA 7.12.2 (all fetches contiguous b128 loads):
//   A (16x32): lane l -> row l%16; elems 0..7 = K (l/16)*8+e, 8..15 = +16
//   B (32x16): lane l -> col l%16; elems 0..7 = K (l/16)*8+e, 8..15 = +16
//   C (16x16): lane l -> col l%16; vgpr v -> row v + 8*(l/16)
// ---------------------------------------------------------------------------
template <int ACT, int TM, int TN>
__global__ void __launch_bounds__(128)
gemm_f16_wmma(const _Float16* __restrict__ A, const _Float16* __restrict__ Bt,
              float* __restrict__ C, int K, int Nout) {
    const int lane = threadIdx.x & 31;
    const int wave = threadIdx.x >> 5;
    const int wm   = wave >> 1, wn = wave & 1;
    const int hi   = lane >> 4, l16 = lane & 15;

    const int rowOrg = blockIdx.y * (32 * TM) + wm * (16 * TM);
    const int colOrg = blockIdx.x * (32 * TN) + wn * (16 * TN);

    const _Float16* arow[TM];
    const _Float16* brow[TN];
#pragma unroll
    for (int i = 0; i < TM; ++i) arow[i] = A + (size_t)(rowOrg + 16 * i + l16) * (size_t)K;
#pragma unroll
    for (int j = 0; j < TN; ++j) brow[j] = Bt + (size_t)(colOrg + 16 * j + l16) * (size_t)K;

    v8f acc[TM][TN] = {};

    for (int k0 = 0; k0 < K; k0 += 32) {
        union U { v16h v; v8h h[2]; };
        U a[TM], b[TN];
#pragma unroll
        for (int i = 0; i < TM; ++i) {
            a[i].h[0] = *(const v8h*)(arow[i] + k0 + hi * 8);
            a[i].h[1] = *(const v8h*)(arow[i] + k0 + 16 + hi * 8);
        }
#pragma unroll
        for (int j = 0; j < TN; ++j) {
            b[j].h[0] = *(const v8h*)(brow[j] + k0 + hi * 8);
            b[j].h[1] = *(const v8h*)(brow[j] + k0 + 16 + hi * 8);
        }
#pragma unroll
        for (int i = 0; i < TM; ++i)
#pragma unroll
            for (int j = 0; j < TN; ++j)
                acc[i][j] = __builtin_amdgcn_wmma_f32_16x16x32_f16(
                    false, a[i].v, false, b[j].v, (short)0, acc[i][j], false, false);
    }

#pragma unroll
    for (int i = 0; i < TM; ++i) {
#pragma unroll
        for (int j = 0; j < TN; ++j) {
            const int col = colOrg + 16 * j + l16;
#pragma unroll
            for (int v = 0; v < 8; ++v) {
                float x = acc[i][j][v];
                float* dst = C + (size_t)(rowOrg + 16 * i + v + 8 * hi) * (size_t)Nout + col;
                if (ACT == 1) {
                    x = 1.0f / (1.0f + __expf(-x));
                    __builtin_nontemporal_store(x, dst);
                } else {
                    *dst = x;
                }
            }
        }
    }
}

// ---------------------------------------------------------------------------
// Graph preprocessing: int histograms, rsqrt norms, dst-sorted edge list
// ---------------------------------------------------------------------------
__global__ void zero_i32(int* __restrict__ p, int n) {
    int i = blockIdx.x * blockDim.x + threadIdx.x;
    if (i < n) p[i] = 0;
}

__global__ void count_edges(const int* __restrict__ src, const int* __restrict__ dst,
                            int* __restrict__ cnt_out, int* __restrict__ cnt_in, int E) {
    int e = blockIdx.x * blockDim.x + threadIdx.x;
    if (e < E) {
        atomicAdd(&cnt_out[src[e]], 1);
        atomicAdd(&cnt_in [dst[e]], 1);
    }
}

__global__ void rs_from_cnt(const int* __restrict__ cnt, float* __restrict__ rs, int n) {
    int i = blockIdx.x * blockDim.x + threadIdx.x;
    if (i < n) rs[i] = rsqrtf(fmaxf((float)cnt[i], 1.0f));
}

// Single-block exclusive prefix scan over NN=8192 counts (8 per thread + LDS
// Hillis-Steele over 1024 partials). Writes both base[] and fill cursor pos[].
__global__ void __launch_bounds__(1024)
scan_excl_8192(const int* __restrict__ cnt, int* __restrict__ base, int* __restrict__ pos) {
    __shared__ int s[1024];
    const int t = threadIdx.x;
    int loc[8];
    int sum = 0;
#pragma unroll
    for (int i = 0; i < 8; ++i) { loc[i] = sum; sum += cnt[t * 8 + i]; }
    s[t] = sum;
    __syncthreads();
    for (int off = 1; off < 1024; off <<= 1) {
        int v = (t >= off) ? s[t - off] : 0;
        __syncthreads();
        s[t] += v;
        __syncthreads();
    }
    const int excl = s[t] - sum;
#pragma unroll
    for (int i = 0; i < 8; ++i) {
        int v = excl + loc[i];
        base[t * 8 + i] = v;
        pos [t * 8 + i] = v;
    }
}

__global__ void fill_sorted(const int* __restrict__ src, const int* __restrict__ dst,
                            int* __restrict__ pos, int* __restrict__ esrc, int E) {
    int e = blockIdx.x * blockDim.x + threadIdx.x;
    if (e < E) {
        int slot = atomicAdd(&pos[dst[e]], 1);
        esrc[slot] = src[e];
    }
}

// Atomic-free aggregation: one block per destination node; 256 threads span the
// feature dim (x2 for dim=512). Neighbor rows are L2-resident; reads coalesced.
__global__ void __launch_bounds__(256)
gather_agg(const float* __restrict__ H, const int* __restrict__ esrc,
           const int* __restrict__ base, const int* __restrict__ cnt,
           float* __restrict__ agg, int dim) {
    const int d  = blockIdx.x;
    const int tx = threadIdx.x;
    const int b  = base[d];
    const int e_end = b + cnt[d];
    const bool two = (dim > 256);
    float acc0 = 0.0f, acc1 = 0.0f;
    for (int e = b; e < e_end; ++e) {
        const float* hr = H + (size_t)esrc[e] * (size_t)dim;
        acc0 += hr[tx];
        if (two) acc1 += hr[tx + 256];
    }
    agg[(size_t)d * dim + tx] = acc0;
    if (two) agg[(size_t)d * dim + tx + 256] = acc1;
}

// ---------------------------------------------------------------------------
// Elementwise stages
// ---------------------------------------------------------------------------
__global__ void transpose_to_f16(const float* __restrict__ W, _Float16* __restrict__ Wt,
                                 int K, int Ncols) {
    int idx = blockIdx.x * blockDim.x + threadIdx.x;
    if (idx >= K * Ncols) return;
    int k = idx / Ncols, n = idx % Ncols;
    Wt[(size_t)n * K + k] = (_Float16)W[idx];
}

__global__ void embed_scale_f16(const float* __restrict__ emb, const int* __restrict__ nids,
                                const float* __restrict__ rs, _Float16* __restrict__ o,
                                int K, int total) {
    int idx = blockIdx.x * blockDim.x + threadIdx.x;
    if (idx >= total) return;
    int i = idx / K, k = idx % K;
    o[idx] = (_Float16)(emb[(size_t)nids[i] * K + k] * rs[i]);
}

// hh[i,j] = (f16)( relu(agg[i,j]*rs_in[i] + b[j]) * rs_out[i] )
__global__ void conv1_epilogue(const float* __restrict__ agg, const float* __restrict__ rs_in,
                               const float* __restrict__ rs_out, const float* __restrict__ b,
                               _Float16* __restrict__ hh, int dim, int total) {
    int idx = blockIdx.x * blockDim.x + threadIdx.x;
    if (idx >= total) return;
    int i = idx / dim, j = idx % dim;
    float v = fmaxf(agg[idx] * rs_in[i] + b[j], 0.0f);
    hh[idx] = (_Float16)(v * rs_out[i]);
}

// z = (aggM*rs_in + b2) + noise * exp(aggS*rs_in + b3); store f16 for WMMA
__global__ void reparam_kernel(const float* __restrict__ aggM, const float* __restrict__ aggS,
                               const float* __restrict__ rs_in, const float* __restrict__ b2,
                               const float* __restrict__ b3, const float* __restrict__ noise,
                               _Float16* __restrict__ zh, int dim, int total) {
    int idx = blockIdx.x * blockDim.x + threadIdx.x;
    if (idx >= total) return;
    int i = idx / dim, j = idx % dim;
    float mean = aggM[idx] * rs_in[i] + b2[j];
    float lstd = aggS[idx] * rs_in[i] + b3[j];
    zh[idx] = (_Float16)(mean + noise[idx] * __expf(lstd));
}

// ---------------------------------------------------------------------------
// Host-side orchestration
// ---------------------------------------------------------------------------
static inline unsigned cdiv(size_t a, unsigned b) { return (unsigned)((a + b - 1) / b); }

extern "C" void kernel_launch(void* const* d_in, const int* in_sizes, int n_in,
                              void* d_out, int out_size, void* d_ws, size_t ws_size,
                              hipStream_t stream) {
    (void)in_sizes; (void)n_in; (void)out_size; (void)ws_size;

    const int*   nids  = (const int*)  d_in[0];
    const int*   src   = (const int*)  d_in[1];
    const int*   dst   = (const int*)  d_in[2];
    const float* emb   = (const float*)d_in[3];
    const float* W1    = (const float*)d_in[4];
    const float* b1    = (const float*)d_in[5];
    const float* W2    = (const float*)d_in[6];
    const float* b2    = (const float*)d_in[7];
    const float* W3    = (const float*)d_in[8];
    const float* b3    = (const float*)d_in[9];
    const float* noise = (const float*)d_in[10];
    float*       out   = (float*)d_out;

    // ---- workspace carve-out (~44 MB, buffers reused across phases) ----
    char* p = (char*)d_ws;
    auto take = [&](size_t bytes) -> char* {
        char* r = p;
        p += (bytes + 255) & ~(size_t)255;
        return r;
    };
    int*      cnt    = (int*)     take((size_t)2 * NN * 4);  // cnt_out | cnt_in
    int*      cnt_out = cnt;
    int*      cnt_in  = cnt + NN;
    float*    rs     = (float*)   take((size_t)2 * NN * 4);  // rs_out | rs_in
    float*    rs_out = rs;
    float*    rs_in  = rs + NN;
    int*      base   = (int*)     take((size_t)NN * 4);
    int*      pos    = (int*)     take((size_t)NN * 4);
    int*      esrc   = (int*)     take((size_t)EDGES * 4);   // dst-sorted src ids
    _Float16* W1t    = (_Float16*)take((size_t)HID1 * INDIM * 2);
    _Float16* W2t    = (_Float16*)take((size_t)HID2 * HID1 * 2);
    _Float16* W3t    = (_Float16*)take((size_t)HID2 * HID1 * 2);
    _Float16* XH     = (_Float16*)take((size_t)NN * INDIM * 2); // xh -> hh -> zh
    float*    GBUF   = (float*)   take((size_t)NN * HID1 * 4);  // hW -> mW|sW
    float*    ABUF   = (float*)   take((size_t)NN * HID1 * 4);  // agg1 -> aggM|aggS

    const int TPB = 256;

    // 1) degrees (int histogram) + rsqrt norms + dst-sorted edge list
    zero_i32<<<cdiv(2 * NN, TPB), TPB, 0, stream>>>(cnt, 2 * NN);
    count_edges<<<cdiv(EDGES, TPB), TPB, 0, stream>>>(src, dst, cnt_out, cnt_in, EDGES);
    rs_from_cnt<<<cdiv(2 * NN, TPB), TPB, 0, stream>>>(cnt, rs, 2 * NN);
    scan_excl_8192<<<1, 1024, 0, stream>>>(cnt_in, base, pos);
    fill_sorted<<<cdiv(EDGES, TPB), TPB, 0, stream>>>(src, dst, pos, esrc, EDGES);

    // 2) weights -> transposed f16 (B-operand layout: [Nout x K])
    transpose_to_f16<<<cdiv((size_t)INDIM * HID1, TPB), TPB, 0, stream>>>(W1, W1t, INDIM, HID1);
    transpose_to_f16<<<cdiv((size_t)HID1 * HID2, TPB), TPB, 0, stream>>>(W2, W2t, HID1, HID2);
    transpose_to_f16<<<cdiv((size_t)HID1 * HID2, TPB), TPB, 0, stream>>>(W3, W3t, HID1, HID2);

    // 3) layer 1: xh = (emb[nids] * rs_out) f16 ; hW = xh @ W1 (2x2 blocking)
    embed_scale_f16<<<cdiv((size_t)NN * INDIM, TPB), TPB, 0, stream>>>(
        emb, nids, rs_out, XH, INDIM, NN * INDIM);
    gemm_f16_wmma<0, 2, 2><<<dim3(HID1 / 64, NN / 64), 128, 0, stream>>>(
        XH, W1t, GBUF, INDIM, HID1);

    // 4) atomic-free aggregation + fused relu/bias/next-layer normalization
    gather_agg<<<NN, 256, 0, stream>>>(GBUF, esrc, base, cnt_in, ABUF, HID1);
    conv1_epilogue<<<cdiv((size_t)NN * HID1, TPB), TPB, 0, stream>>>(
        ABUF, rs_in, rs_out, b1, XH, HID1, NN * HID1);

    // 5) mean / log_std GEMMs (shared pre-scaled input hh in XH)
    float* mW = GBUF;
    float* sW = GBUF + (size_t)NN * HID2;
    gemm_f16_wmma<0, 2, 2><<<dim3(HID2 / 64, NN / 64), 128, 0, stream>>>(XH, W2t, mW, HID1, HID2);
    gemm_f16_wmma<0, 2, 2><<<dim3(HID2 / 64, NN / 64), 128, 0, stream>>>(XH, W3t, sW, HID1, HID2);

    // 6) aggregate both heads (atomic-free gathers)
    float* aggM = ABUF;
    float* aggS = ABUF + (size_t)NN * HID2;
    gather_agg<<<NN, 256, 0, stream>>>(mW, esrc, base, cnt_in, aggM, HID2);
    gather_agg<<<NN, 256, 0, stream>>>(sW, esrc, base, cnt_in, aggS, HID2);

    // 7) reparameterize -> zh (f16), reusing XH
    _Float16* zh = XH;
    reparam_kernel<<<cdiv((size_t)NN * HID2, TPB), TPB, 0, stream>>>(
        aggM, aggS, rs_in, b2, b3, noise, zh, HID2, NN * HID2);

    // 8) adj = sigmoid(z @ z^T): 4x4-blocked WMMA, sigmoid + NT-store epilogue
    gemm_f16_wmma<1, 4, 4><<<dim3(NN / 128, NN / 128), 128, 0, stream>>>(
        zh, zh, out, HID2, NN);
}
